// RegGNN_8546984919612
// MI455X (gfx1250) — compile-verified, matrix operand order checked by reading, not run on hardware
//
#include <hip/hip_runtime.h>
#include <hip/hip_bf16.h>
#include <math.h>

// ---------------- problem constants (match reference) ----------------
constexpr int NN  = 50000;           // nodes
constexpr int EE  = 800000;          // edges (before self loops)
constexpr int E2  = EE + NN;         // edges incl self loops
constexpr int DIN = 256;             // input feature dim
constexpr int HID = 128;             // H*C = 2*64

typedef __attribute__((ext_vector_type(16))) _Float16 v16h;
typedef __attribute__((ext_vector_type(8)))  float    v8f;

// ---------------- small helpers ----------------
__device__ inline void atomicMaxF32(float* addr, float val) {
    int cur = __float_as_int(*addr);
    while (__int_as_float(cur) < val) {
        int assumed = cur;
        cur = atomicCAS((int*)addr, assumed, __float_as_int(val));
        if (cur == assumed) break;
    }
}

// ---------------- WMMA GEMM:  Y[n,128] = X[n,K] @ W[K,128] + bias ----------------
// block = 256 threads = 8 waves; wave w owns column tile w (16 cols); block covers 64 rows.
// Both A (activations, staged per 16-row tile) and B (weights, staged once) live in LDS
// as f16 so every wave feeds v_wmma_f32_16x16x32_f16 from ds_load_b128 pairs.
template<int K>
__global__ __launch_bounds__(256)
void gemm_bias_wmma(const float* __restrict__ X, const float* __restrict__ W,
                    const float* __restrict__ bias, float* __restrict__ Y, int nrows)
{
    __shared__ _Float16 Wt[128 * K];          // Wt[col*K + k]
    __shared__ _Float16 At[16 * K];           // At[r*K + k], r = row within tile
    const int tid = threadIdx.x;
    for (int i = tid; i < K * 128; i += 256) {        // coalesced read of W[k,128]
        int k = i >> 7, col = i & 127;
        Wt[col * K + k] = (_Float16)W[i];
    }

    const int lane = tid & 31;
    const int wave = tid >> 5;
    const int col  = wave * 16 + (lane & 15);
    const int hi   = lane >> 4;                        // half-wave selector
    const int blockRow0 = blockIdx.x * 64;

    const _Float16* arow = &At[(size_t)(lane & 15) * K + hi * 16];
    const _Float16* brow = &Wt[(size_t)col * K + hi * 16];

    for (int rt = 0; rt < 4; ++rt) {
        const int row0 = blockRow0 + rt * 16;
        if (row0 >= nrows) break;
        __syncthreads();   // Wt ready (1st iter) / previous At consumers done
        // stage 16 x K activation tile as f16 (coalesced float4 reads)
        constexpr int K4 = K / 4;
        for (int i4 = tid; i4 < 16 * K4; i4 += 256) {
            int r = i4 / K4, k = (i4 % K4) * 4;
            float4 f = *(const float4*)(X + (size_t)(row0 + r) * K + k);
            _Float16* dst = &At[(size_t)r * K + k];
            dst[0] = (_Float16)f.x; dst[1] = (_Float16)f.y;
            dst[2] = (_Float16)f.z; dst[3] = (_Float16)f.w;
        }
        __syncthreads();

        v8f c = {};
        #pragma unroll
        for (int kb = 0; kb < K / 32; ++kb) {
            v16h a = *(const v16h*)(arow + kb * 32);   // 16 contiguous halves (32B)
            v16h b = *(const v16h*)(brow + kb * 32);
            c = __builtin_amdgcn_wmma_f32_16x16x32_f16(false, a, false, b,
                                                       (short)0, c, false, false);
        }
        const float bs = bias[col];
        #pragma unroll
        for (int r = 0; r < 8; ++r) {                  // D: row = row0 + 8*hi + r, col
            Y[(size_t)(row0 + hi * 8 + r) * 128 + col] = c[r] + bs;
        }
    }
}

// ---------------- column statistics: sum / sumsq over rows ----------------
__global__ __launch_bounds__(256)
void colstats_kernel(const float* __restrict__ X, int nrows, int dcols,
                     float* __restrict__ sum, float* __restrict__ sumsq)
{
    const int tid  = threadIdx.x;
    const int col  = tid % dcols;
    const int rsub = tid / dcols;
    const int rpi  = 256 / dcols;                      // rows per block-iter (1 or 2)
    float s = 0.f, s2 = 0.f;
    for (int r = blockIdx.x * rpi + rsub; r < nrows; r += gridDim.x * rpi) {
        float v = X[(size_t)r * dcols + col];
        s += v; s2 += v * v;
    }
    atomicAdd(&sum[col], s);
    atomicAdd(&sumsq[col], s2);
}

// graph_norm apply: y = w*(x - mean*ms)/sqrt(var+1e-5)+b with var of (x-mean*ms)
__global__ __launch_bounds__(256)
void gn_apply_kernel(const float* __restrict__ X, const float* __restrict__ sum,
                     const float* __restrict__ sumsq, const float* __restrict__ w,
                     const float* __restrict__ b, const float* __restrict__ ms,
                     float* __restrict__ Y, int nrows, int dcols)
{
    size_t i = (size_t)blockIdx.x * 256 + threadIdx.x;
    size_t tot = (size_t)nrows * dcols;
    if (i >= tot) return;
    int col = (int)(i % dcols);
    float mean = sum[col] / (float)nrows;
    float t    = mean * ms[col];
    float var  = sumsq[col] / (float)nrows - 2.f * t * mean + t * t;
    Y[i] = w[col] * (X[i] - t) * rsqrtf(var + 1e-5f) + b[col];
}

// batch-norm apply + relu (in place)
__global__ __launch_bounds__(256)
void bn_apply_relu_kernel(float* __restrict__ X, const float* __restrict__ sum,
                          const float* __restrict__ sumsq, const float* __restrict__ g,
                          const float* __restrict__ b, int nrows, int dcols)
{
    size_t i = (size_t)blockIdx.x * 256 + threadIdx.x;
    size_t tot = (size_t)nrows * dcols;
    if (i >= tot) return;
    int col = (int)(i % dcols);
    float mean = sum[col] / (float)nrows;
    float var  = sumsq[col] / (float)nrows - mean * mean;
    float v = g[col] * (X[i] - mean) * rsqrtf(var + 1e-5f) + b[col];
    X[i] = v > 0.f ? v : 0.f;
}

// ---------------- reductions / inits ----------------
__global__ __launch_bounds__(256)
void reduce_sum_kernel(const float* __restrict__ v, int n, float* __restrict__ out)
{
    __shared__ float sm[256];
    float s = 0.f;
    for (int i = blockIdx.x * 256 + threadIdx.x; i < n; i += gridDim.x * 256) s += v[i];
    sm[threadIdx.x] = s; __syncthreads();
    for (int st = 128; st > 0; st >>= 1) {
        if (threadIdx.x < st) sm[threadIdx.x] += sm[threadIdx.x + st];
        __syncthreads();
    }
    if (threadIdx.x == 0) atomicAdd(out, sm[0]);
}

__global__ __launch_bounds__(256)
void lmax_init_kernel(float* __restrict__ lmax, int n)
{
    int i = blockIdx.x * 256 + threadIdx.x;
    if (i < n) lmax[i] = -INFINITY;
}

__global__ __launch_bounds__(256)
void lmax_fix_kernel(float* __restrict__ lmax, float* __restrict__ denom, int n)
{
    int i = blockIdx.x * 256 + threadIdx.x;
    if (i >= n) return;
    float v = lmax[i];
    lmax[i] = (v == v && v > -3.0e38f && v < 3.0e38f) ? v : 0.f;  // isfinite
    denom[i] = 0.f;
}

// ---------------- edge kernels: one wave per edge; lane l -> channels 4l..4l+3 ----------------
// lanes 0..15 = head 0 (c 0..63), lanes 16..31 = head 1 (c 64..127)
__global__ __launch_bounds__(256)
void edge_logits_kernel(const int* __restrict__ ei, const float* __restrict__ ew,
                        const float* __restrict__ ewsum,
                        const float* __restrict__ xl, const float* __restrict__ xr,
                        const float* __restrict__ We, const float* __restrict__ att,
                        float* __restrict__ elog, float* __restrict__ lmax)
{
    const int e = blockIdx.x * 8 + (threadIdx.x >> 5);
    const int lane = threadIdx.x & 31;
    if (e >= E2) return;
    int s, d; float w;
    if (e < EE) { s = ei[e]; d = ei[EE + e]; w = ew[e]; }
    else        { s = d = e - EE; w = ewsum[0] / (float)EE; }
    const float4 a  = ((const float4*)(xl + (size_t)s * 128))[lane];
    const float4 r  = ((const float4*)(xr + (size_t)d * 128))[lane];
    const float4 we = ((const float4*)We)[lane];
    const float4 at = ((const float4*)att)[lane];
    float m0 = a.x + r.x + w * we.x; m0 = m0 > 0.f ? m0 : 0.2f * m0;
    float m1 = a.y + r.y + w * we.y; m1 = m1 > 0.f ? m1 : 0.2f * m1;
    float m2 = a.z + r.z + w * we.z; m2 = m2 > 0.f ? m2 : 0.2f * m2;
    float m3 = a.w + r.w + w * we.w; m3 = m3 > 0.f ? m3 : 0.2f * m3;
    float p = m0 * at.x + m1 * at.y + m2 * at.z + m3 * at.w;
    for (int mask = 1; mask < 16; mask <<= 1) p += __shfl_xor(p, mask, 32);
    if ((lane & 15) == 0) {
        int h = lane >> 4;
        elog[(size_t)e * 2 + h] = p;
        atomicMaxF32(&lmax[(size_t)d * 2 + h], p);
    }
}

__global__ __launch_bounds__(256)
void edge_exp_kernel(const int* __restrict__ ei, float* __restrict__ elog,
                     const float* __restrict__ lmax, float* __restrict__ denom)
{
    int e = blockIdx.x * 256 + threadIdx.x;
    if (e >= E2) return;
    int d = (e < EE) ? ei[EE + e] : (e - EE);
    float x0 = expf(elog[(size_t)e * 2 + 0] - lmax[(size_t)d * 2 + 0]);
    float x1 = expf(elog[(size_t)e * 2 + 1] - lmax[(size_t)d * 2 + 1]);
    elog[(size_t)e * 2 + 0] = x0;
    elog[(size_t)e * 2 + 1] = x1;
    atomicAdd(&denom[(size_t)d * 2 + 0], x0);
    atomicAdd(&denom[(size_t)d * 2 + 1], x1);
}

__global__ __launch_bounds__(256)
void edge_accum_kernel(const int* __restrict__ ei, const float* __restrict__ elog,
                       const float* __restrict__ denom, const float* __restrict__ xl,
                       float* __restrict__ agg)
{
    const int e = blockIdx.x * 8 + (threadIdx.x >> 5);
    const int lane = threadIdx.x & 31;
    if (e >= E2) return;
    int s, d;
    if (e < EE) { s = ei[e]; d = ei[EE + e]; } else { s = d = e - EE; }
    float a0 = elog[(size_t)e * 2 + 0] / (denom[(size_t)d * 2 + 0] + 1e-16f);
    float a1 = elog[(size_t)e * 2 + 1] / (denom[(size_t)d * 2 + 1] + 1e-16f);
    float al = (lane < 16) ? a0 : a1;
    float4 v = ((const float4*)(xl + (size_t)s * 128))[lane];
    float* o = agg + (size_t)d * 128 + lane * 4;
    atomicAdd(o + 0, al * v.x);
    atomicAdd(o + 1, al * v.y);
    atomicAdd(o + 2, al * v.z);
    atomicAdd(o + 3, al * v.w);
}

__global__ __launch_bounds__(256)
void bias_relu_kernel(const float* __restrict__ agg, const float* __restrict__ b,
                      float* __restrict__ out, int nrows)
{
    size_t i = (size_t)blockIdx.x * 256 + threadIdx.x;
    if (i >= (size_t)nrows * 128) return;
    float v = agg[i] + b[i & 127];
    out[i] = v > 0.f ? v : 0.f;
}

// ---------------- fc2: out[n] = dot(h[n,:], W2) + b2 ; one wave per node ----------------
__global__ __launch_bounds__(256)
void fc2_kernel(const float* __restrict__ h, const float* __restrict__ W2,
                const float* __restrict__ b2, float* __restrict__ out, int n)
{
    const int node = blockIdx.x * 8 + (threadIdx.x >> 5);
    const int lane = threadIdx.x & 31;
    if (node >= n) return;
    float4 a = ((const float4*)(h + (size_t)node * 128))[lane];
    float4 w = ((const float4*)W2)[lane];
    float p = a.x * w.x + a.y * w.y + a.z * w.z + a.w * w.w;
    for (int m = 1; m < 32; m <<= 1) p += __shfl_xor(p, m, 32);
    if (lane == 0) out[node] = p + b2[0];
}

// ---------------- host-side orchestration ----------------
static void run_gat_edges(const int* ei, const float* ew, const float* ewsum,
                          const float* xl, const float* xr,
                          const float* We, const float* att, const float* bgat,
                          float* elog, float* lmax, float* denom,
                          float* agg, float* hout, hipStream_t stream)
{
    const int WB = (E2 + 7) / 8;        // wave-per-edge blocks
    const int TB = (E2 + 255) / 256;    // thread-per-edge blocks
    lmax_init_kernel<<<(NN * 2 + 255) / 256, 256, 0, stream>>>(lmax, NN * 2);
    edge_logits_kernel<<<WB, 256, 0, stream>>>(ei, ew, ewsum, xl, xr, We, att, elog, lmax);
    lmax_fix_kernel<<<(NN * 2 + 255) / 256, 256, 0, stream>>>(lmax, denom, NN * 2);
    edge_exp_kernel<<<TB, 256, 0, stream>>>(ei, elog, lmax, denom);
    hipMemsetAsync(agg, 0, (size_t)NN * 128 * sizeof(float), stream);
    edge_accum_kernel<<<WB, 256, 0, stream>>>(ei, elog, denom, xl, agg);
    bias_relu_kernel<<<(NN * 128 + 255) / 256, 256, 0, stream>>>(agg, bgat, hout, NN);
}

extern "C" void kernel_launch(void* const* d_in, const int* in_sizes, int n_in,
                              void* d_out, int out_size, void* d_ws, size_t ws_size,
                              hipStream_t stream)
{
    // inputs in setup_inputs() dict insertion order (params recursed in order)
    const float* x      = (const float*)d_in[0];
    const int*   ei     = (const int*)  d_in[1];
    const float* ew     = (const float*)d_in[2];
    const float* gn1_w  = (const float*)d_in[3];
    const float* gn1_b  = (const float*)d_in[4];
    const float* gn1_ms = (const float*)d_in[5];
    const float* g1_Wl  = (const float*)d_in[6];
    const float* g1_bl  = (const float*)d_in[7];
    const float* g1_Wr  = (const float*)d_in[8];
    const float* g1_br  = (const float*)d_in[9];
    const float* g1_We  = (const float*)d_in[10];
    const float* g1_att = (const float*)d_in[11];
    const float* g1_b   = (const float*)d_in[12];
    const float* g2_Wl  = (const float*)d_in[13];
    const float* g2_bl  = (const float*)d_in[14];
    const float* g2_Wr  = (const float*)d_in[15];
    const float* g2_br  = (const float*)d_in[16];
    const float* g2_We  = (const float*)d_in[17];
    const float* g2_att = (const float*)d_in[18];
    const float* g2_b   = (const float*)d_in[19];
    const float* gn2_w  = (const float*)d_in[20];
    const float* gn2_b  = (const float*)d_in[21];
    const float* gn2_ms = (const float*)d_in[22];
    const float* fc1_W  = (const float*)d_in[23];
    const float* fc1_b  = (const float*)d_in[24];
    const float* bn_g   = (const float*)d_in[25];
    const float* bn_b   = (const float*)d_in[26];
    const float* fc2_W  = (const float*)d_in[27];
    const float* fc2_b  = (const float*)d_in[28];
    float* out = (float*)d_out;

    // workspace layout (floats)
    float* ws = (float*)d_ws;
    size_t offA  = 0;                                  // N*256: h0; later agg (lo) + h2 (hi)
    size_t offXL = (size_t)NN * 256;                   // N*128
    size_t offXR = offXL + (size_t)NN * 128;           // N*128
    size_t offH  = offXR + (size_t)NN * 128;           // N*128: h1 / fc1 out
    size_t offEL = offH  + (size_t)NN * 128;           // E2*2
    size_t offLM = offEL + (size_t)E2 * 2;             // N*2
    size_t offDN = offLM + (size_t)NN * 2;             // N*2
    size_t offST = offDN + (size_t)NN * 2;             // 1024 stats
    float* A    = ws + offA;
    float* agg  = A;                                   // N*128 (reuse of A)
    float* h2   = A + (size_t)NN * 128;                // N*128 (upper half of A)
    float* XL   = ws + offXL;
    float* XR   = ws + offXR;
    float* Hb   = ws + offH;
    float* EL   = ws + offEL;
    float* LM   = ws + offLM;
    float* DN   = ws + offDN;
    float* S    = ws + offST;                          // [0..255]=sum [256..511]=sumsq [512]=ewsum

    const int GEMM_BLOCKS = (NN + 63) / 64;

    // 0) stats reset + edge-weight sum + graph_norm 1
    hipMemsetAsync(S, 0, 1024 * sizeof(float), stream);
    reduce_sum_kernel<<<256, 256, 0, stream>>>(ew, EE, &S[512]);
    colstats_kernel<<<256, 256, 0, stream>>>(x, NN, DIN, S, S + 256);
    gn_apply_kernel<<<(int)(((size_t)NN * DIN + 255) / 256), 256, 0, stream>>>(
        x, S, S + 256, gn1_w, gn1_b, gn1_ms, A, NN, DIN);

    // 1) GAT layer 1 (K=256)
    gemm_bias_wmma<256><<<GEMM_BLOCKS, 256, 0, stream>>>(A, g1_Wl, g1_bl, XL, NN);
    gemm_bias_wmma<256><<<GEMM_BLOCKS, 256, 0, stream>>>(A, g1_Wr, g1_br, XR, NN);
    run_gat_edges(ei, ew, &S[512], XL, XR, g1_We, g1_att, g1_b, EL, LM, DN, agg, Hb, stream);

    // 2) GAT layer 2 (K=128)
    gemm_bias_wmma<128><<<GEMM_BLOCKS, 256, 0, stream>>>(Hb, g2_Wl, g2_bl, XL, NN);
    gemm_bias_wmma<128><<<GEMM_BLOCKS, 256, 0, stream>>>(Hb, g2_Wr, g2_br, XR, NN);
    run_gat_edges(ei, ew, &S[512], XL, XR, g2_We, g2_att, g2_b, EL, LM, DN, agg, h2, stream);

    // 3) graph_norm 2 : h2 -> agg region (reuse)
    hipMemsetAsync(S, 0, 1024 * sizeof(float), stream);
    colstats_kernel<<<256, 256, 0, stream>>>(h2, NN, HID, S, S + 256);
    gn_apply_kernel<<<(int)(((size_t)NN * HID + 255) / 256), 256, 0, stream>>>(
        h2, S, S + 256, gn2_w, gn2_b, gn2_ms, agg, NN, HID);

    // 4) fc1 (K=128) : agg -> Hb
    gemm_bias_wmma<128><<<GEMM_BLOCKS, 256, 0, stream>>>(agg, fc1_W, fc1_b, Hb, NN);

    // 5) batch-norm + relu in place on Hb
    hipMemsetAsync(S, 0, 1024 * sizeof(float), stream);
    colstats_kernel<<<256, 256, 0, stream>>>(Hb, NN, HID, S, S + 256);
    bn_apply_relu_kernel<<<(int)(((size_t)NN * HID + 255) / 256), 256, 0, stream>>>(
        Hb, S, S + 256, bn_g, bn_b, NN, HID);

    // 6) fc2 -> out
    fc2_kernel<<<(NN + 7) / 8, 256, 0, stream>>>(Hb, fc2_W, fc2_b, out, NN);

    (void)in_sizes; (void)n_in; (void)out_size; (void)ws_size;
}
